// GCN_MUTAG_53824530153627
// MI455X (gfx1250) — matrix operand order repaired and checked by reading, not compiled, not run on hardware
//
#include <hip/hip_runtime.h>

typedef __attribute__((ext_vector_type(16))) _Float16 v16h;
typedef __attribute__((ext_vector_type(8)))  _Float16 h8;
typedef __attribute__((ext_vector_type(2)))  _Float16 h2;
typedef __attribute__((ext_vector_type(8)))  float    v8f;

#define BM 128
#define BN 64
#define BK 32

// ---------------------------------------------------------------------------
// Branch-free WMMA f16 GEMM on pre-packed, pre-padded operands.
//   A  : [Mp][K]  f16 row-major (K = lda, multiple of 32)
//   Bt : [Np][K]  f16 row-major = B transposed (so tiles copy as b128)
//   C  : [Mp][ldc] f32 or f16
// 256 threads = 8 wave32; block tile 128x64; each wave owns 4 16x16 C tiles
// (4 v_wmma per K-step sharing one B fragment).
// ---------------------------------------------------------------------------
__global__ __launch_bounds__(256) void gemm_f16_wmma(
    const _Float16* __restrict__ A, const _Float16* __restrict__ Bt,
    void* __restrict__ Cout, int K, int ldc, int storeF32)
{
    __shared__ _Float16 As[BM * BK];   // 8 KB
    __shared__ _Float16 Bs[BN * BK];   // 4 KB

    const int tid  = threadIdx.x;
    const int lane = tid & 31;
    const int wave = tid >> 5;             // 0..7
    const int wrp  = (wave >> 2) * 16;     // 0 or 16
    const int wcol = (wave & 3) * 16;      // 0,16,32,48

    const int blockM = blockIdx.y * BM;
    const int blockN = blockIdx.x * BN;

    const int m = lane & 15;
    const int n = lane & 15;
    const int koffA = (lane < 16) ? 0 : 16;   // A: contiguous k = 16*(lane/16) + s, same as B
    const int koffB = (lane < 16) ? 0 : 16;   // B: contiguous k = 16*(lane/16) + s, same as A

    v8f acc[4] = {{}, {}, {}, {}};

    const int srow = tid >> 2;            // 0..63
    const int scol = (tid & 3) * 8;       // 0,8,16,24  (16B aligned)

    const _Float16* Aptr0 = A  + (size_t)(blockM + srow)      * K + scol;
    const _Float16* Aptr1 = A  + (size_t)(blockM + srow + 64) * K + scol;
    const _Float16* Bptr  = Bt + (size_t)(blockN + srow)      * K + scol;

    for (int k0 = 0; k0 < K; k0 += BK) {
        // stage A (128x32) and Bt (64x32): pure b128 copies, no guards
        *(h8*)&As[srow        * BK + scol] = *(const h8*)(Aptr0 + k0);
        *(h8*)&As[(srow + 64) * BK + scol] = *(const h8*)(Aptr1 + k0);
        *(h8*)&Bs[srow        * BK + scol] = *(const h8*)(Bptr  + k0);
        if (k0 + BK < K) {
            __builtin_prefetch(Aptr0 + k0 + BK, 0, 0);
            __builtin_prefetch(Aptr1 + k0 + BK, 0, 0);
            __builtin_prefetch(Bptr  + k0 + BK, 0, 0);
        }
        __syncthreads();

        // B fragment: 16 contiguous halves per lane -> 2x ds_load_b128
        h8 bl = *(const h8*)&Bs[(wcol + n) * BK + koffB];
        h8 bh = *(const h8*)&Bs[(wcol + n) * BK + koffB + 8];
        v16h b = __builtin_shufflevector(bl, bh,
                 0,1,2,3,4,5,6,7,8,9,10,11,12,13,14,15);

#pragma unroll
        for (int t = 0; t < 4; ++t) {
            const int rbase = (wrp + t * 32 + m) * BK;
            h8 al = *(const h8*)&As[rbase + koffA];        // K = koffA+0..7
            h8 ah = *(const h8*)&As[rbase + koffA + 8];    // K = koffA+8..15
            v16h a = __builtin_shufflevector(al, ah,
                     0,1,2,3,4,5,6,7,8,9,10,11,12,13,14,15);
            acc[t] = __builtin_amdgcn_wmma_f32_16x16x32_f16(
                false, a, false, b, (short)0, acc[t], false, false);
        }
        __syncthreads();
    }

    // D layout: VGPR r -> M = r (lanes 0-15) / r+8 (lanes 16-31), N = lane&15
    const int mbase = (lane < 16) ? 0 : 8;
#pragma unroll
    for (int t = 0; t < 4; ++t) {
#pragma unroll
        for (int r = 0; r < 8; ++r) {
            size_t gm = (size_t)blockM + wrp + t * 32 + mbase + r;
            size_t gn = (size_t)blockN + wcol + n;
            if (storeF32) ((float*)Cout)[gm * ldc + gn] = acc[t][r];
            else ((_Float16*)Cout)[gm * ldc + gn] = (_Float16)acc[t][r];
        }
    }
}

// ---------------------------------------------------------------------------
// Packing kernels (fp32 -> padded f16)
// ---------------------------------------------------------------------------
__global__ void pack_pad_f16(const float* __restrict__ src, _Float16* __restrict__ dst,
                             int R, int Ccols, int Rp, int Cp)
{
    long i = blockIdx.x * (long)blockDim.x + threadIdx.x;
    if (i >= (long)Rp * Cp) return;
    int r = (int)(i / Cp), c = (int)(i % Cp);
    float v = (r < R && c < Ccols) ? src[(size_t)r * Ccols + c] : 0.0f;
    dst[i] = (_Float16)v;
}

// W [K][N] fp32 -> Wt [Np][Kp] f16 (transposed + padded)
__global__ void pack_wt_f16(const float* __restrict__ W, _Float16* __restrict__ Wt,
                            int K, int N, int Kp, int Np)
{
    long i = blockIdx.x * (long)blockDim.x + threadIdx.x;
    if (i >= (long)Np * Kp) return;
    int nn = (int)(i / Kp), k = (int)(i % Kp);
    float v = (k < K && nn < N) ? W[(size_t)k * N + nn] : 0.0f;
    Wt[i] = (_Float16)v;
}

// ---------------------------------------------------------------------------
// Degree / normalization:  dinv = rsqrt(1 + indegree)
// ---------------------------------------------------------------------------
__global__ void deg_init(float* deg, int n) {
    int i = blockIdx.x * blockDim.x + threadIdx.x;
    if (i < n) deg[i] = 1.0f;
}
__global__ void deg_count(float* deg, const int* __restrict__ dst, int E) {
    int i = blockIdx.x * blockDim.x + threadIdx.x;
    if (i < E) atomicAdd(&deg[dst[i]], 1.0f);
}
__global__ void deg_rsqrt(float* deg, int n) {
    int i = blockIdx.x * blockDim.x + threadIdx.x;
    if (i < n) deg[i] = rsqrtf(deg[i]);
}

// ---------------------------------------------------------------------------
// Edge aggregation: agg[dst] += h[src] * dinv[src]*dinv[dst]  (+ self loops)
// One wave per (edge|node); lanes stride Hp/2 f16 pairs (exact, branch-free).
// h pad columns are zero, so they add nothing.
// ---------------------------------------------------------------------------
__global__ void aggregate_edges(const _Float16* __restrict__ h, float* __restrict__ agg,
                                const int* __restrict__ src, const int* __restrict__ dst,
                                const float* __restrict__ dinv, int E, int Nn, int Hp)
{
    int gwave = (int)((blockIdx.x * (long)blockDim.x + threadIdx.x) >> 5);
    int lane  = threadIdx.x & 31;
    if (gwave >= E + Nn) return;
    int s, d;
    if (gwave < E) { s = src[gwave]; d = dst[gwave]; }
    else           { s = d = gwave - E; }
    float norm = dinv[s] * dinv[d];
    const _Float16* hr = h + (size_t)s * Hp;
    float* ar = agg + (size_t)d * Hp;
    const int P = Hp >> 1;
    for (int p = lane; p < P; p += 32) {
        h2 v = *(const h2*)&hr[2 * p];
        atomicAdd(&ar[2 * p],     (float)v[0] * norm);
        atomicAdd(&ar[2 * p + 1], (float)v[1] * norm);
    }
}

// agg f32 [Mp][Hp] + bias -> relu -> f16 [Mp][Hp], pads zeroed (writes all)
__global__ void bias_relu_pack(const float* __restrict__ agg, const float* __restrict__ b,
                               _Float16* __restrict__ out, int rowsReal, int H,
                               int Mp, int Hp)
{
    long i = blockIdx.x * (long)blockDim.x + threadIdx.x;
    if (i >= (long)Mp * Hp) return;
    int r = (int)(i / Hp), c = (int)(i % Hp);
    float v = 0.0f;
    if (r < rowsReal && c < H) {
        v = agg[i] + b[c];
        v = v > 0.0f ? v : 0.0f;
    }
    out[i] = (_Float16)v;
}

// segment_max; post-ReLU inputs >= 0 so uint-bit atomicMax is order-preserving
// and 0-init matches the reference's isfinite->0 cleanup.
__global__ void pool_max(const _Float16* __restrict__ h, const int* __restrict__ batch,
                         unsigned* __restrict__ pooled, int Nn, int H, int Hp)
{
    long i = blockIdx.x * (long)blockDim.x + threadIdx.x;
    if (i >= (long)Nn * H) return;
    int node = (int)(i / H), f = (int)(i % H);
    int g = batch[node];
    float v = (float)h[(size_t)node * Hp + f];
    atomicMax(&pooled[(size_t)g * H + f], __float_as_uint(v));
}

// small linear head (C classes) + log_softmax, one thread per graph
__global__ void final_logits(const _Float16* __restrict__ z, const float* __restrict__ Wl2,
                             const float* __restrict__ bl2, float* __restrict__ out,
                             int G, int H, int Hp, int C)
{
    int g = blockIdx.x * blockDim.x + threadIdx.x;
    if (g >= G) return;
    float acc[8];
    for (int c = 0; c < C; ++c) acc[c] = bl2[c];
    const _Float16* zr = z + (size_t)g * Hp;
    for (int f = 0; f < H; ++f) {
        float v = (float)zr[f];
        for (int c = 0; c < C; ++c) acc[c] += v * Wl2[(size_t)f * C + c];
    }
    float mx = acc[0];
    for (int c = 1; c < C; ++c) mx = fmaxf(mx, acc[c]);
    float s = 0.0f;
    for (int c = 0; c < C; ++c) s += expf(acc[c] - mx);
    float lse = mx + logf(s);
    for (int c = 0; c < C; ++c) out[(size_t)g * C + c] = acc[c] - lse;
}

// ---------------------------------------------------------------------------
extern "C" void kernel_launch(void* const* d_in, const int* in_sizes, int n_in,
                              void* d_out, int out_size, void* d_ws, size_t ws_size,
                              hipStream_t stream)
{
    (void)n_in; (void)ws_size;
    const float* x   = (const float*)d_in[0];
    const int* eidx  = (const int*)d_in[1];
    const int* batch = (const int*)d_in[2];
    const float* W1  = (const float*)d_in[3];
    const float* b1  = (const float*)d_in[4];
    const float* W2  = (const float*)d_in[5];
    const float* b2  = (const float*)d_in[6];
    const float* W3  = (const float*)d_in[7];
    const float* b3  = (const float*)d_in[8];
    const float* Wl1 = (const float*)d_in[9];
    const float* bl1 = (const float*)d_in[10];
    const float* Wl2 = (const float*)d_in[11];
    const float* bl2 = (const float*)d_in[12];
    float* out = (float*)d_out;

    const int Nn  = in_sizes[2];            // 50000 nodes
    const int Fin = in_sizes[0] / Nn;       // 128
    const int E   = in_sizes[1] / 2;        // 800000
    const int H   = in_sizes[4];            // 418
    const int C   = in_sizes[12];           // 2
    const int G   = out_size / C;           // 418 graphs

    const int Mp  = ((Nn + BM - 1) / BM) * BM;   // 50048
    const int Hp  = ((H + BN - 1) / BN) * BN;    // 448  (also mult of 32 -> K pad)
    const int Kp1 = ((Fin + BK - 1) / BK) * BK;  // 128
    const int Gp  = ((G + BM - 1) / BM) * BM;    // 512

    const int* src = eidx;
    const int* dst = eidx + E;

    // workspace layout
    char* ws = (char*)d_ws;
    size_t off = 0;
    auto alloc = [&](size_t bytes) { char* p = ws + off; off += (bytes + 255) & ~(size_t)255; return p; };
    float*    dinv = (float*)   alloc((size_t)Nn * 4);
    _Float16* Ah   = (_Float16*)alloc((size_t)Mp * Hp * 2);  // activations (f16, padded)
    _Float16* Gh   = (_Float16*)alloc((size_t)Mp * Hp * 2);  // gemm f16 out / pooledh
    float*    Agg  = (float*)   alloc((size_t)Mp * Hp * 4);  // aggregation accum / head f32
    _Float16* Wt1  = (_Float16*)alloc((size_t)Hp * Kp1 * 2);
    _Float16* Wt2  = (_Float16*)alloc((size_t)Hp * Hp * 2);
    _Float16* Wt3  = (_Float16*)alloc((size_t)Hp * Hp * 2);
    _Float16* Wtl1 = (_Float16*)alloc((size_t)Hp * Hp * 2);

    const int aggBlocks = (int)(((long)E + Nn + 7) / 8);       // 8 waves / block
    const int mhBlocks  = (int)(((long)Mp * Hp + 255) / 256);
    const int nhBlocks  = (int)(((long)Nn * H + 255) / 256);

    dim3 gridConv(Hp / BN, Mp / BM);
    dim3 gridHead(Hp / BN, Gp / BM);

    // ---- pack weights (transposed + padded, f16) ----
    pack_wt_f16<<<(int)(((long)Hp * Kp1 + 255) / 256), 256, 0, stream>>>(W1, Wt1, Fin, H, Kp1, Hp);
    pack_wt_f16<<<(int)(((long)Hp * Hp  + 255) / 256), 256, 0, stream>>>(W2, Wt2, H, H, Hp, Hp);
    pack_wt_f16<<<(int)(((long)Hp * Hp  + 255) / 256), 256, 0, stream>>>(W3, Wt3, H, H, Hp, Hp);
    pack_wt_f16<<<(int)(((long)Hp * Hp  + 255) / 256), 256, 0, stream>>>(Wl1, Wtl1, H, H, Hp, Hp);

    // ---- degree normalization ----
    deg_init <<<(Nn + 255) / 256, 256, 0, stream>>>(dinv, Nn);
    deg_count<<<(E  + 255) / 256, 256, 0, stream>>>(dinv, dst, E);
    deg_rsqrt<<<(Nn + 255) / 256, 256, 0, stream>>>(dinv, Nn);

    // ---- pack x -> Ah [Mp][Kp1] ----
    pack_pad_f16<<<(int)(((long)Mp * Kp1 + 255) / 256), 256, 0, stream>>>(x, Ah, Nn, Fin, Mp, Kp1);

    // ---- layer 1 ----
    gemm_f16_wmma<<<gridConv, 256, 0, stream>>>(Ah, Wt1, Gh, Kp1, Hp, 0);
    hipMemsetAsync(Agg, 0, (size_t)Mp * Hp * 4, stream);
    aggregate_edges<<<aggBlocks, 256, 0, stream>>>(Gh, Agg, src, dst, dinv, E, Nn, Hp);
    bias_relu_pack<<<mhBlocks, 256, 0, stream>>>(Agg, b1, Ah, Nn, H, Mp, Hp);

    // ---- layer 2 ----
    gemm_f16_wmma<<<gridConv, 256, 0, stream>>>(Ah, Wt2, Gh, Hp, Hp, 0);
    hipMemsetAsync(Agg, 0, (size_t)Mp * Hp * 4, stream);
    aggregate_edges<<<aggBlocks, 256, 0, stream>>>(Gh, Agg, src, dst, dinv, E, Nn, Hp);
    bias_relu_pack<<<mhBlocks, 256, 0, stream>>>(Agg, b2, Ah, Nn, H, Mp, Hp);

    // ---- layer 3 ----
    gemm_f16_wmma<<<gridConv, 256, 0, stream>>>(Ah, Wt3, Gh, Hp, Hp, 0);
    hipMemsetAsync(Agg, 0, (size_t)Mp * Hp * 4, stream);
    aggregate_edges<<<aggBlocks, 256, 0, stream>>>(Gh, Agg, src, dst, dinv, E, Nn, Hp);
    bias_relu_pack<<<mhBlocks, 256, 0, stream>>>(Agg, b3, Ah, Nn, H, Mp, Hp);

    // ---- global max pool: Ah (h3, >=0) -> pooled f32 [G][H] (reuse Agg) ----
    unsigned* pooled = (unsigned*)Agg;
    hipMemsetAsync(pooled, 0, (size_t)G * H * 4, stream);
    pool_max<<<nhBlocks, 256, 0, stream>>>(Ah, batch, pooled, Nn, H, Hp);

    // ---- head: z = relu(pooled @ Wl1 + bl1); out = log_softmax(z @ Wl2 + bl2)
    pack_pad_f16<<<(int)(((long)Gp * Hp + 255) / 256), 256, 0, stream>>>(
        (const float*)pooled, Gh, G, H, Gp, Hp);
    float* zf32 = Agg;  // pooled already consumed (stream-ordered)
    gemm_f16_wmma<<<gridHead, 256, 0, stream>>>(Gh, Wtl1, zf32, Hp, Hp, 1);
    bias_relu_pack<<<(int)(((long)Gp * Hp + 255) / 256), 256, 0, stream>>>(
        zf32, bl1, Ah, G, H, Gp, Hp);
    final_logits<<<(G + 127) / 128, 128, 0, stream>>>(Ah, Wl2, bl2, out, G, H, Hp, C);
}
